// TxtFlagLargeDiT_17841294147685
// MI455X (gfx1250) — matrix operand order repaired and checked by reading, not compiled, not run on hardware
//
#include <hip/hip_runtime.h>
#include <hip/hip_bf16.h>
#include <math.h>
#include <stdint.h>

typedef __attribute__((ext_vector_type(16))) __bf16 v16bf;
typedef __attribute__((ext_vector_type(8)))  __bf16 v8bf;
typedef __attribute__((ext_vector_type(4)))  __bf16 v4bf;
typedef __attribute__((ext_vector_type(8)))  float  v8f;
typedef long long i64;

#define BMT 64
#define BNT 64
#define BKT 32
#define LSTR 40   // padded LDS row stride in halves (32 data + 8 pad)

// ---------- WMMA fragment loads from LDS (layouts per CDNA5 ISA) ----------
// A 16x32 bf16: lanes 0-15 hold M=lane, K{0..7,16..23}; lanes 16-31 hold M=lane-16, K{8..15,24..31}
__device__ __forceinline__ v16bf frag_a(const __bf16* lds, int r0, int lane) {
  int lr = lane & 15, hi = (lane >> 4) & 1;
  const __bf16* p = lds + (r0 + lr) * LSTR + hi * 8;
  union { v16bf v; v8bf h[2]; } u;
  u.h[0] = *(const v8bf*)(p);
  u.h[1] = *(const v8bf*)(p + 16);
  return u.v;
}
// B 32x16 bf16 staged as [N][K]: lanes 0-15 hold N=lane, K{0..7,16..23}; lanes 16-31 hold N=lane-16, K{8..15,24..31}
__device__ __forceinline__ v16bf frag_b(const __bf16* lds, int c0, int lane) {
  int lr = lane & 15, hi = (lane >> 4) & 1;
  const __bf16* p = lds + (c0 + lr) * LSTR + hi * 8;
  union { v16bf v; v8bf h[2]; } u;
  u.h[0] = *(const v8bf*)(p);
  u.h[1] = *(const v8bf*)(p + 16);
  return u.v;
}

// ---------- staging: global fp32 -> registers (float4), then registers -> LDS bf16 ----------
// Each thread owns quads i = tid and tid+256 of the 512 quads in a 64x32 tile.
__device__ __forceinline__ void load_pair(const float* __restrict__ G, int g0, int rows_max,
                                          int ld, int k0, int K, int tid, float4* r) {
  if ((g0 + 64 <= rows_max) && (k0 + BKT <= K)) {
    // interior fast path: unguarded 16B vector loads
#pragma unroll
    for (int t = 0; t < 2; ++t) {
      int i = tid + t * 256;
      int row = i >> 3, cg = (i & 7) << 2;
      r[t] = *(const float4*)(G + (i64)(g0 + row) * ld + (k0 + cg));
    }
  } else {
#pragma unroll
    for (int t = 0; t < 2; ++t) {
      int i = tid + t * 256;
      int row = i >> 3, cg = (i & 7) << 2;
      int gr = g0 + row, gk = k0 + cg;
      float4 v = make_float4(0.f, 0.f, 0.f, 0.f);
      if (gr < rows_max) {
        const float* p = G + (i64)gr * ld + gk;
        if (gk + 0 < K) v.x = p[0];
        if (gk + 1 < K) v.y = p[1];
        if (gk + 2 < K) v.z = p[2];
        if (gk + 3 < K) v.w = p[3];
      }
      r[t] = v;
    }
  }
}

__device__ __forceinline__ void scale_pair(float4* r, float s0, float s1) {
  r[0].x *= s0; r[0].y *= s0; r[0].z *= s0; r[0].w *= s0;
  r[1].x *= s1; r[1].y *= s1; r[1].z *= s1; r[1].w *= s1;
}

__device__ __forceinline__ void store_lds(__bf16* lds, int tid, const float4* r) {
#pragma unroll
  for (int t = 0; t < 2; ++t) {
    int i = tid + t * 256;
    int row = i >> 3, cg = (i & 7) << 2;
    v4bf d;
    d.x = (__bf16)r[t].x; d.y = (__bf16)r[t].y; d.z = (__bf16)r[t].z; d.w = (__bf16)r[t].w;
    *(v4bf*)(lds + row * LSTR + cg) = d;
  }
}

// ---------- Generic batched NT GEMM:  C[z](MxN) (+)= A[z](MxK) * B[z](NxK)^T + bias ----------
__global__ void __launch_bounds__(256)
gemm_nt(const float* __restrict__ A, const float* __restrict__ B,
        const float* __restrict__ bias, const float* __restrict__ rowScale,
        float* __restrict__ C,
        int M, int N, int K, int lda, int ldb, int ldc, int zmod,
        i64 sA1, i64 sA2, i64 sB1, i64 sB2, i64 sC1, i64 sC2,
        int transC, int accum)
{
  __shared__ __attribute__((aligned(16))) __bf16 sAt[BMT * LSTR];
  __shared__ __attribute__((aligned(16))) __bf16 sBt[BNT * LSTR];
  int z = blockIdx.z; int zb = z / zmod; int zr = z - zb * zmod;
  A += zb * sA1 + zr * sA2;
  B += zb * sB1 + zr * sB2;
  C += zb * sC1 + zr * sC2;
  int m0 = blockIdx.x * BMT, n0 = blockIdx.y * BNT;
  int tid = threadIdx.x, lane = tid & 31, w = tid >> 5;
  int wr0 = (w & 3) * 16, wc0 = (w >> 2) * 32;
  // per-thread A-row gate scales are fixed across the whole K loop: hoist them
  bool hasScale = (rowScale != nullptr);
  float s0 = 1.f, s1 = 1.f;
  if (hasScale) {
    int r0i = m0 + (tid >> 3), r1i = m0 + ((tid + 256) >> 3);
    s0 = (r0i < M) ? rowScale[r0i] : 0.f;
    s1 = (r1i < M) ? rowScale[r1i] : 0.f;
  }
  v8f acc0 = {}; v8f acc1 = {};
  float4 rA[2], rB[2], nA[2], nB[2];
  load_pair(A, m0, M, lda, 0, K, tid, rA);
  load_pair(B, n0, N, ldb, 0, K, tid, rB);
  if (hasScale) scale_pair(rA, s0, s1);
  int k0 = 0;
  for (; k0 + BKT < K; k0 += BKT) {   // steady state: branch-free prefetch
    store_lds(sAt, tid, rA);
    store_lds(sBt, tid, rB);
    __syncthreads();
    load_pair(A, m0, M, lda, k0 + BKT, K, tid, nA);
    load_pair(B, n0, N, ldb, k0 + BKT, K, tid, nB);
    if (hasScale) scale_pair(nA, s0, s1);
    v16bf af = frag_a(sAt, wr0, lane);
    v16bf b0 = frag_b(sBt, wc0, lane);
    v16bf b1 = frag_b(sBt, wc0 + 16, lane);
    acc0 = __builtin_amdgcn_wmma_f32_16x16x32_bf16(false, af, false, b0, (short)0, acc0, false, false);
    acc1 = __builtin_amdgcn_wmma_f32_16x16x32_bf16(false, af, false, b1, (short)0, acc1, false, false);
    __syncthreads();
    rA[0] = nA[0]; rA[1] = nA[1]; rB[0] = nB[0]; rB[1] = nB[1];
  }
  // peeled final tile
  store_lds(sAt, tid, rA);
  store_lds(sBt, tid, rB);
  __syncthreads();
  {
    v16bf af = frag_a(sAt, wr0, lane);
    v16bf b0 = frag_b(sBt, wc0, lane);
    v16bf b1 = frag_b(sBt, wc0 + 16, lane);
    acc0 = __builtin_amdgcn_wmma_f32_16x16x32_bf16(false, af, false, b0, (short)0, acc0, false, false);
    acc1 = __builtin_amdgcn_wmma_f32_16x16x32_bf16(false, af, false, b1, (short)0, acc1, false, false);
  }
  int lr = lane & 15, hi = (lane >> 4) & 1;
  for (int j = 0; j < 2; ++j) {
    v8f a = j ? acc1 : acc0;
    int n = n0 + wc0 + j * 16 + lr;
    if (n >= N) continue;
    float bv = bias ? bias[n] : 0.0f;
#pragma unroll
    for (int r = 0; r < 8; ++r) {
      int m = m0 + wr0 + r + hi * 8;
      if (m >= M) continue;
      float val = a[r] + bv;
      float* p = transC ? (C + (i64)n * ldc + m) : (C + (i64)m * ldc + n);
      if (accum) *p += val; else *p = val;
    }
  }
}

// ---------- SwiGLU up-projection: H[z] = silu(A*B1^T) .* (A*B3^T) ----------
__global__ void __launch_bounds__(256)
gemm_swiglu_nt(const float* __restrict__ A, const float* __restrict__ B1,
               const float* __restrict__ B3, float* __restrict__ H,
               int M, int N, int K, int lda, int ldb, int ldh,
               i64 sAz, i64 sB1z, i64 sB3z, i64 sHz)
{
  __shared__ __attribute__((aligned(16))) __bf16 sAt[BMT * LSTR];
  __shared__ __attribute__((aligned(16))) __bf16 sB1t[BNT * LSTR];
  __shared__ __attribute__((aligned(16))) __bf16 sB3t[BNT * LSTR];
  int z = blockIdx.z;
  A += (i64)z * sAz; B1 += (i64)z * sB1z; B3 += (i64)z * sB3z; H += (i64)z * sHz;
  int m0 = blockIdx.x * BMT, n0 = blockIdx.y * BNT;
  int tid = threadIdx.x, lane = tid & 31, w = tid >> 5;
  int wr0 = (w & 3) * 16, wc0 = (w >> 2) * 32;
  v8f a1_0 = {}; v8f a1_1 = {}; v8f a3_0 = {}; v8f a3_1 = {};
  float4 rA[2], rB1[2], rB3[2], nA[2], nB1[2], nB3[2];
  load_pair(A, m0, M, lda, 0, K, tid, rA);
  load_pair(B1, n0, N, ldb, 0, K, tid, rB1);
  load_pair(B3, n0, N, ldb, 0, K, tid, rB3);
  int k0 = 0;
  for (; k0 + BKT < K; k0 += BKT) {
    store_lds(sAt, tid, rA);
    store_lds(sB1t, tid, rB1);
    store_lds(sB3t, tid, rB3);
    __syncthreads();
    load_pair(A, m0, M, lda, k0 + BKT, K, tid, nA);
    load_pair(B1, n0, N, ldb, k0 + BKT, K, tid, nB1);
    load_pair(B3, n0, N, ldb, k0 + BKT, K, tid, nB3);
    v16bf af  = frag_a(sAt, wr0, lane);
    v16bf b10 = frag_b(sB1t, wc0, lane);
    v16bf b11 = frag_b(sB1t, wc0 + 16, lane);
    v16bf b30 = frag_b(sB3t, wc0, lane);
    v16bf b31 = frag_b(sB3t, wc0 + 16, lane);
    a1_0 = __builtin_amdgcn_wmma_f32_16x16x32_bf16(false, af, false, b10, (short)0, a1_0, false, false);
    a1_1 = __builtin_amdgcn_wmma_f32_16x16x32_bf16(false, af, false, b11, (short)0, a1_1, false, false);
    a3_0 = __builtin_amdgcn_wmma_f32_16x16x32_bf16(false, af, false, b30, (short)0, a3_0, false, false);
    a3_1 = __builtin_amdgcn_wmma_f32_16x16x32_bf16(false, af, false, b31, (short)0, a3_1, false, false);
    __syncthreads();
    rA[0] = nA[0]; rA[1] = nA[1];
    rB1[0] = nB1[0]; rB1[1] = nB1[1];
    rB3[0] = nB3[0]; rB3[1] = nB3[1];
  }
  store_lds(sAt, tid, rA);
  store_lds(sB1t, tid, rB1);
  store_lds(sB3t, tid, rB3);
  __syncthreads();
  {
    v16bf af  = frag_a(sAt, wr0, lane);
    v16bf b10 = frag_b(sB1t, wc0, lane);
    v16bf b11 = frag_b(sB1t, wc0 + 16, lane);
    v16bf b30 = frag_b(sB3t, wc0, lane);
    v16bf b31 = frag_b(sB3t, wc0 + 16, lane);
    a1_0 = __builtin_amdgcn_wmma_f32_16x16x32_bf16(false, af, false, b10, (short)0, a1_0, false, false);
    a1_1 = __builtin_amdgcn_wmma_f32_16x16x32_bf16(false, af, false, b11, (short)0, a1_1, false, false);
    a3_0 = __builtin_amdgcn_wmma_f32_16x16x32_bf16(false, af, false, b30, (short)0, a3_0, false, false);
    a3_1 = __builtin_amdgcn_wmma_f32_16x16x32_bf16(false, af, false, b31, (short)0, a3_1, false, false);
  }
  int lr = lane & 15, hi = (lane >> 4) & 1;
  for (int j = 0; j < 2; ++j) {
    v8f h1 = j ? a1_1 : a1_0;
    v8f h3 = j ? a3_1 : a3_0;
    int n = n0 + wc0 + j * 16 + lr;
    if (n >= N) continue;
#pragma unroll
    for (int r = 0; r < 8; ++r) {
      int m = m0 + wr0 + r + hi * 8;
      if (m >= M) continue;
      float x1 = h1[r];
      float sl = x1 * (1.0f / (1.0f + __expf(-x1)));
      H[(i64)m * ldh + n] = sl * h3[r];
    }
  }
}

// ---------- softmax over rows of 128 (attention scores), scaled by 1/sqrt(144) ----------
__global__ void softmax128(float* __restrict__ s, int rows) {
  int gw = (blockIdx.x * blockDim.x + threadIdx.x) >> 5;
  int lane = threadIdx.x & 31;
  if (gw >= rows) return;
  float* p = s + (i64)gw * 128;
  const float scale = 1.0f / 12.0f;
  float v[4], mx = -3.4e38f;
#pragma unroll
  for (int i = 0; i < 4; ++i) { v[i] = p[lane + 32 * i] * scale; mx = fmaxf(mx, v[i]); }
  for (int m = 16; m > 0; m >>= 1) mx = fmaxf(mx, __shfl_xor(mx, m));
  float sum = 0.f;
#pragma unroll
  for (int i = 0; i < 4; ++i) { v[i] = __expf(v[i] - mx); sum += v[i]; }
  for (int m = 16; m > 0; m >>= 1) sum += __shfl_xor(sum, m);
  float inv = 1.0f / sum;
#pragma unroll
  for (int i = 0; i < 4; ++i) p[lane + 32 * i] = v[i] * inv;
}

// ---------- gumbel RNG (deterministic hash; JAX threefry not reproducible) ----------
__device__ __forceinline__ uint64_t mix64(uint64_t x) {
  x += 0x9E3779B97F4A7C15ull;
  x = (x ^ (x >> 30)) * 0xBF58476D1CE4E5B9ull;
  x = (x ^ (x >> 27)) * 0x94D049BB133111EBull;
  return x ^ (x >> 31);
}
__device__ __forceinline__ float gumbelf(uint32_t strm, uint32_t n, uint32_t j) {
  uint64_t h = mix64(((uint64_t)strm << 40) ^ ((uint64_t)n << 8) ^ (uint64_t)j);
  float u = ((float)(uint32_t)(h >> 40) + 0.5f) * (1.0f / 16777216.0f);
  return -__logf(-__logf(u));
}

// ---------- gating: one wave per token ----------
__global__ void __launch_bounds__(256)
gates_kernel(const float* __restrict__ ca, const float* __restrict__ ac,
             const float* __restrict__ tme,
             const float* __restrict__ hlw, const float* __restrict__ hlb,
             const float* __restrict__ cgw, const float* __restrict__ cgb,
             const float* __restrict__ agw, const float* __restrict__ agb,
             float* __restrict__ wsc, float* __restrict__ acc, int Ntok)
{
  int gw = (blockIdx.x * blockDim.x + threadIdx.x) >> 5;
  int lane = threadIdx.x & 31;
  if (gw >= Ntok) return;
  int b = gw / 1024;
  const float* pca = ca + (i64)gw * 1152;
  const float* pac = ac + (i64)gw * 1152;
  const float* pt  = tme + (i64)b * 1152;
  float h0 = 0.f, h1 = 0.f;
  float c[4] = {0.f, 0.f, 0.f, 0.f};
  float a[4] = {0.f, 0.f, 0.f, 0.f};
  for (int i = lane; i < 1152; i += 32) {
    float tca = pca[i], tac = pac[i], tt = pt[i];
    h0 += tt * hlw[i];
    h1 += tt * hlw[1152 + i];
#pragma unroll
    for (int e = 0; e < 4; ++e) {
      c[e] += tca * cgw[e * 1152 + i];
      a[e] += tac * agw[e * 1152 + i];
    }
  }
  for (int m = 16; m > 0; m >>= 1) {
    h0 += __shfl_xor(h0, m); h1 += __shfl_xor(h1, m);
#pragma unroll
    for (int e = 0; e < 4; ++e) { c[e] += __shfl_xor(c[e], m); a[e] += __shfl_xor(a[e], m); }
  }
  if (lane != 0) return;
  // soft gumbel-softmax on hl (temp 1.0)
  float l0 = h0 + hlb[0] + gumbelf(1u, (uint32_t)gw, 0u);
  float l1 = h1 + hlb[1] + gumbelf(1u, (uint32_t)gw, 1u);
  float mx = fmaxf(l0, l1);
  float e0 = __expf(l0 - mx), e1 = __expf(l1 - mx);
  float cm = e0 / (e0 + e1), am = e1 / (e0 + e1);
  // hard gumbel one-hot on cap / ac gates (temp 2.0; argmax is temp-invariant)
  int argc = 0, arga = 0;
  float bc = -3.4e38f, ba = -3.4e38f;
#pragma unroll
  for (int e = 0; e < 4; ++e) {
    float lc = c[e] + cgb[e] + gumbelf(2u, (uint32_t)gw, (uint32_t)e);
    float la = a[e] + agb[e] + gumbelf(3u, (uint32_t)gw, (uint32_t)e);
    if (lc > bc) { bc = lc; argc = e; }
    if (la > ba) { ba = la; arga = e; }
  }
#pragma unroll
  for (int e = 0; e < 4; ++e) {
    float cpe = (e == argc) ? 1.0f : 0.0f;
    float ape = (e == arga) ? 1.0f : 0.0f;
    wsc[e * Ntok + gw]       = cpe * cm;   // MoE down-proj row scale (cap)
    wsc[(4 + e) * Ntok + gw] = ape * am;   // MoE down-proj row scale (ac)
    atomicAdd(&acc[e], cpe * cm);
    atomicAdd(&acc[4 + e], ape * am);
  }
  atomicAdd(&acc[8], 4.0f * (cm + am));
}

__global__ void loss_finalize(const float* __restrict__ acc, float* __restrict__ out) {
  if (threadIdx.x == 0 && blockIdx.x == 0) {
    float denom = acc[8] + 1e-10f;
    float s = 0.f;
#pragma unroll
    for (int j = 0; j < 8; ++j) {
      float u = acc[j] / denom;
      s += u * __logf(u + 1e-10f);
    }
    out[0] = s * (1.0f / 8.0f);
  }
}

extern "C" void kernel_launch(void* const* d_in, const int* in_sizes, int n_in,
                              void* d_out, int out_size, void* d_ws, size_t ws_size,
                              hipStream_t stream) {
  (void)in_sizes; (void)n_in; (void)ws_size;
  const int D = 1152, E = 4, HID = 768, NH = 8, Bb = 2, S = 1024, Sc = 128, HD = 144;
  const int Ntok = Bb * S; // 2048
  const float* x    = (const float*)d_in[0];
  const float* tme  = (const float*)d_in[1];
  const float* cap  = (const float*)d_in[2];
  const float* acst = (const float*)d_in[3];
  const float* aiw  = (const float*)d_in[4];
  const float* aib  = (const float*)d_in[5];
  const float* aow  = (const float*)d_in[6];
  const float* aob  = (const float*)d_in[7];
  const float* hlw  = (const float*)d_in[8];
  const float* hlb  = (const float*)d_in[9];
  const float* cgw  = (const float*)d_in[10];
  const float* cgb  = (const float*)d_in[11];
  const float* agw  = (const float*)d_in[12];
  const float* agb  = (const float*)d_in[13];
  const float* cw1  = (const float*)d_in[14];
  const float* cw2  = (const float*)d_in[15];
  const float* cw3  = (const float*)d_in[16];
  const float* aw1  = (const float*)d_in[17];
  const float* aw2  = (const float*)d_in[18];
  const float* aw3  = (const float*)d_in[19];
  const float* fw1  = (const float*)d_in[20];
  const float* fw2  = (const float*)d_in[21];
  const float* fw3  = (const float*)d_in[22];
  float* zout = (float*)d_out;

  // workspace layout (floats)
  float* ws = (float*)d_ws;
  i64 o = 0;
  float* q   = ws + o; o += (i64)Ntok * D;          // 2048x1152
  float* kp  = ws + o; o += (i64)Bb * Sc * D;       // 256x1152
  float* vT  = ws + o; o += (i64)Bb * D * Sc;       // [b][d][sc]
  float* sc  = ws + o; o += (i64)Bb * NH * S * Sc;  // scores/probs
  float* ao  = ws + o; o += (i64)Ntok * D;          // attention output
  float* ca  = ws + o; o += (i64)Ntok * D;          // after out-proj
  float* Hb  = ws + o; o += (i64)E * Ntok * HID;    // expert hidden (reused)
  float* y   = ws + o; o += (i64)Ntok * D;          // zc+za
  float* wsc = ws + o; o += 8LL * Ntok;             // per-token expert scales
  float* acc = ws + o; o += 16;                     // loss accumulators

  dim3 blk(256);
  // --- MHA projections ---
  gemm_nt<<<dim3(32, 18, 1), blk, 0, stream>>>(x, aiw, aib, nullptr, q,
      Ntok, D, D, D, D, D, 1, 0, 0, 0, 0, 0, 0, 0, 0);
  gemm_nt<<<dim3(4, 18, 1), blk, 0, stream>>>(cap, aiw + (i64)D * D, aib + D, nullptr, kp,
      Bb * Sc, D, D, D, D, D, 1, 0, 0, 0, 0, 0, 0, 0, 0);
  // V projection, stored transposed per batch: vT[b][d][sc]
  gemm_nt<<<dim3(2, 18, Bb), blk, 0, stream>>>(cap, aiw + 2LL * D * D, aib + 2 * D, nullptr, vT,
      Sc, D, D, D, D, /*ldc*/Sc, 1, (i64)Sc * D, 0, 0, 0, (i64)D * Sc, 0, /*transC*/1, 0);
  // --- scores: [b][h][q][k] = Q_h K_h^T ---
  gemm_nt<<<dim3(16, 2, Bb * NH), blk, 0, stream>>>(q, kp, nullptr, nullptr, sc,
      S, Sc, HD, D, D, Sc, NH,
      (i64)S * D, HD, (i64)Sc * D, HD, (i64)NH * S * Sc, (i64)S * Sc, 0, 0);
  softmax128<<<dim3((Bb * NH * S) / 8), blk, 0, stream>>>(sc, Bb * NH * S);
  // --- PV: ao[b][q][h*144 + d] ---
  gemm_nt<<<dim3(16, 3, Bb * NH), blk, 0, stream>>>(sc, vT, nullptr, nullptr, ao,
      S, HD, Sc, Sc, Sc, D, NH,
      (i64)NH * S * Sc, (i64)S * Sc, (i64)D * Sc, (i64)HD * Sc, (i64)S * D, HD, 0, 0);
  // --- out projection ---
  gemm_nt<<<dim3(32, 18, 1), blk, 0, stream>>>(ao, aow, aob, nullptr, ca,
      Ntok, D, D, D, D, D, 1, 0, 0, 0, 0, 0, 0, 0, 0);
  // --- gates + loss stats ---
  hipMemsetAsync(acc, 0, 16 * sizeof(float), stream);
  gates_kernel<<<dim3(Ntok / 8), blk, 0, stream>>>(ca, acst, tme, hlw, hlb, cgw, cgb,
      agw, agb, wsc, acc, Ntok);
  // --- cap experts: SwiGLU up (all E batched), then scaled down-proj accumulated into y ---
  gemm_swiglu_nt<<<dim3(32, 12, E), blk, 0, stream>>>(x, cw1, cw3, Hb,
      Ntok, HID, D, D, D, HID, 0, (i64)HID * D, (i64)HID * D, (i64)Ntok * HID);
  for (int e = 0; e < E; ++e)
    gemm_nt<<<dim3(32, 18, 1), blk, 0, stream>>>(Hb + (i64)e * Ntok * HID,
        cw2 + (i64)e * D * HID, nullptr, wsc + (i64)e * Ntok, y,
        Ntok, D, HID, HID, HID, D, 1, 0, 0, 0, 0, 0, 0, 0, (e > 0) ? 1 : 0);
  // --- ac experts (reuse Hb) ---
  gemm_swiglu_nt<<<dim3(32, 12, E), blk, 0, stream>>>(x, aw1, aw3, Hb,
      Ntok, HID, D, D, D, HID, 0, (i64)HID * D, (i64)HID * D, (i64)Ntok * HID);
  for (int e = 0; e < E; ++e)
    gemm_nt<<<dim3(32, 18, 1), blk, 0, stream>>>(Hb + (i64)e * Ntok * HID,
        aw2 + (i64)e * D * HID, nullptr, wsc + (i64)(4 + e) * Ntok, y,
        Ntok, D, HID, HID, HID, D, 1, 0, 0, 0, 0, 0, 0, 0, 1);
  // --- region FFN: per expert e only columns [e*288, (e+1)*288) are live ---
  const int R = D / E; // 288
  gemm_swiglu_nt<<<dim3(32, 12, E), blk, 0, stream>>>(y, fw1, fw3, Hb,
      Ntok, HID, R, D, D, HID, (i64)R, (i64)HID * D + R, (i64)HID * D + R, (i64)Ntok * HID);
  gemm_nt<<<dim3(32, 5, E), blk, 0, stream>>>(Hb, fw2, nullptr, nullptr, zout,
      Ntok, R, HID, HID, HID, D, 1,
      (i64)Ntok * HID, 0, (i64)D * HID + (i64)R * HID, 0, (i64)R, 0, 0, 0);
  // --- lb loss scalar at end of output ---
  loss_finalize<<<1, 1, 0, stream>>>(acc, zout + (out_size - 1));
}
